// HierarchicalMalwareGNN_39058432590506
// MI455X (gfx1250) — matrix operand order repaired and checked by reading, not compiled, not run on hardware
//
#include <hip/hip_runtime.h>

// ---------- CDNA5 WMMA types ----------
typedef __attribute__((ext_vector_type(16))) __bf16          v16bf;
typedef __attribute__((ext_vector_type(16))) unsigned short  v16u;
typedef __attribute__((ext_vector_type(8)))  float           v8f;
typedef __attribute__((ext_vector_type(4)))  unsigned int    v4u;
typedef __attribute__((ext_vector_type(8)))  unsigned int    v8u;

__device__ __forceinline__ unsigned short f2bf(float f) {
  unsigned u = __builtin_bit_cast(unsigned, f);
  u += 0x7FFFu + ((u >> 16) & 1u);          // round-to-nearest-even
  return (unsigned short)(u >> 16);
}

__device__ __forceinline__ void atomicAddF(float* p, float v) {
  __hip_atomic_fetch_add(p, v, __ATOMIC_RELAXED, __HIP_MEMORY_SCOPE_AGENT);
}

__device__ __forceinline__ v16bf combine16(v4u a, v4u b) {
  v8u u = {a[0], a[1], a[2], a[3], b[0], b[1], b[2], b[3]};
  return __builtin_bit_cast(v16bf, u);
}

// A fragment from f32 source (GEMM1 only). ISA 7.12.2 16-bit A layout:
// lane half h: elements 0..7 = K kb+8h..kb+8h+7 ; elements 8..15 = K kb+16+8h..
__device__ __forceinline__ v16bf load_a_frag_f32(const float* __restrict__ A,
                                                 size_t rowOff, int kBase, int half) {
  v16u u;
  const float* p0 = A + rowOff + kBase + half * 8;
  const float* p1 = p0 + 16;
#pragma unroll
  for (int i = 0; i < 8; ++i) {
    u[i]     = f2bf(p0[i]);                 // contiguous: compiler emits b128 pairs
    u[i + 8] = f2bf(p1[i]);
  }
  return __builtin_bit_cast(v16bf, u);
}

// A fragment from pre-converted bf16 row-major source: two contiguous 16B loads.
__device__ __forceinline__ v16bf load_a_frag_bf(const unsigned short* __restrict__ A,
                                                size_t rowOff, int kBase, int half) {
  const v4u* p0 = (const v4u*)(A + rowOff + kBase + half * 8);
  const v4u* p1 = (const v4u*)(A + rowOff + kBase + 16 + half * 8);
  return combine16(*p0, *p1);
}

// B fragment from weights pre-packed in WMMA register order:
// pack index = (((kt*nTilesTot + nt)*32) + lane)*16 bf16 elements (32B per lane).
__device__ __forceinline__ v16bf load_b_packed(const unsigned short* __restrict__ Bp,
                                               int kt, int nt, int nTilesTot, int lane) {
  const v4u* p = (const v4u*)(Bp + ((((size_t)kt * nTilesTot + nt) * 32) + lane) * 16);
  return combine16(p[0], p[1]);
}

__device__ __forceinline__ v8f wmma_bf16(v16bf a, v16bf b, v8f c) {
  return __builtin_amdgcn_wmma_f32_16x16x32_bf16(false, a, false, b, (short)0, c,
                                                 false, false);
}

// ---------- weight packing: f32 [K x N] -> bf16 WMMA B-fragment order ----------
__global__ void pack_b_kernel(const float* __restrict__ B, unsigned short* __restrict__ out,
                              int K, int N) {
  long long i = (long long)blockIdx.x * blockDim.x + threadIdx.x;
  if (i >= (long long)K * N) return;
  int k = (int)(i / N), n = (int)(i % N);
  int kt = k >> 5, kr = k & 31;
  int half = (kr >> 3) & 1, ke = (kr & 7) + ((kr >> 4) << 3);
  int nt = n >> 4, nl = n & 15;
  int nTilesTot = N >> 4;
  size_t dst = ((((size_t)kt * nTilesTot + nt) * 32) + (half * 16 + nl)) * 16 + ke;
  out[dst] = f2bf(B[i]);
}

// ---------- small utility kernels ----------
__global__ void fill_kernel(float* __restrict__ p, float v, long long n) {
  long long i = (long long)blockIdx.x * blockDim.x + threadIdx.x;
  if (i < n) p[i] = v;
}

__global__ void deg_edges_kernel(const int* __restrict__ dst, float* __restrict__ deg, int E) {
  int e = blockIdx.x * blockDim.x + threadIdx.x;
  if (e < E) atomicAddF(&deg[dst[e]], 1.0f);
}

__global__ void rsqrt_kernel(float* __restrict__ p, int n) {
  int i = blockIdx.x * blockDim.x + threadIdx.x;
  if (i < n) p[i] = rsqrtf(p[i]);
}

// relu + bf16 convert (h1 f32 not needed afterwards)
__global__ void relu_cvt_kernel(const float* __restrict__ in, unsigned short* __restrict__ out,
                                long long n) {
  long long i = (long long)blockIdx.x * blockDim.x + threadIdx.x;
  if (i < n) out[i] = f2bf(fmaxf(in[i], 0.0f));
}

// in-place relu + bf16 convert (h2 f32 needed for pooling)
__global__ void relu_inplace_cvt_kernel(float* __restrict__ io, unsigned short* __restrict__ out,
                                        long long n) {
  long long i = (long long)blockIdx.x * blockDim.x + threadIdx.x;
  if (i < n) {
    float v = fmaxf(io[i], 0.0f);
    io[i] = v;
    out[i] = f2bf(v);
  }
}

// ---------- GEMM1: hw1 = x@W1 ; h1_init = hw1*dinv^2 + b1 ----------
__global__ __launch_bounds__(128)
void gemm1_kernel(const float* __restrict__ A, const unsigned short* __restrict__ Bp,
                  const float* __restrict__ bias, const float* __restrict__ dinv,
                  float* __restrict__ rawOut, float* __restrict__ selfOut,
                  int M, int K, int N) {
  int lane = threadIdx.x & 31;
  int wave = blockIdx.x * (blockDim.x >> 5) + (threadIdx.x >> 5);
  int nGroups = N >> 6, nTilesTot = N >> 4;
  int totalWaves = (M >> 4) * nGroups;
  if (wave >= totalWaves) return;            // wave-uniform: EXEC all-ones
  int mt = wave / nGroups, ng = wave % nGroups;
  int rowBase = mt << 4, colBase = ng << 6;
  int half = lane >> 4, nl = lane & 15;
  size_t aRowOff = (size_t)(rowBase + nl) * K;

  v8f acc[4] = {};
  for (int kb = 0; kb < K; kb += 32) {
    v16bf a = load_a_frag_f32(A, aRowOff, kb, half);
    int kt = kb >> 5;
#pragma unroll
    for (int t = 0; t < 4; ++t) {
      v16bf b = load_b_packed(Bp, kt, ng * 4 + t, nTilesTot, lane);
      acc[t] = wmma_bf16(a, b, acc[t]);
    }
  }
#pragma unroll
  for (int t = 0; t < 4; ++t) {
    int col = colBase + t * 16 + nl;
    float bb = bias[col];
#pragma unroll
    for (int r = 0; r < 8; ++r) {
      int row = rowBase + half * 8 + r;      // C layout: VGPR r -> M = r + 8*half
      float c = acc[t][r];
      float di = dinv[row];
      rawOut[(size_t)row * N + col]  = c;
      selfOut[(size_t)row * N + col] = c * di * di + bb;
    }
  }
}

// ---------- GEMM2: msg2 = h1@W2 ; h2_init = h1@Wres + bres + msg2*dinv^2 + b2 ----------
__global__ __launch_bounds__(128)
void gemm2_kernel(const unsigned short* __restrict__ A, const unsigned short* __restrict__ W2p,
                  const unsigned short* __restrict__ Wresp, const float* __restrict__ b2,
                  const float* __restrict__ bres, const float* __restrict__ dinv,
                  float* __restrict__ msg2, float* __restrict__ h2,
                  int M, int K, int N) {
  int lane = threadIdx.x & 31;
  int wave = blockIdx.x * (blockDim.x >> 5) + (threadIdx.x >> 5);
  int nGroups = N >> 6, nTilesTot = N >> 4;
  int totalWaves = (M >> 4) * nGroups;
  if (wave >= totalWaves) return;
  int mt = wave / nGroups, ng = wave % nGroups;
  int rowBase = mt << 4, colBase = ng << 6;
  int half = lane >> 4, nl = lane & 15;
  size_t aRowOff = (size_t)(rowBase + nl) * K;

  v8f acc1[4] = {}, acc2[4] = {};
  for (int kb = 0; kb < K; kb += 32) {
    v16bf a = load_a_frag_bf(A, aRowOff, kb, half);
    int kt = kb >> 5;
#pragma unroll
    for (int t = 0; t < 4; ++t) {
      v16bf bw = load_b_packed(W2p, kt, ng * 4 + t, nTilesTot, lane);
      acc1[t] = wmma_bf16(a, bw, acc1[t]);
      v16bf br = load_b_packed(Wresp, kt, ng * 4 + t, nTilesTot, lane);
      acc2[t] = wmma_bf16(a, br, acc2[t]);
    }
  }
#pragma unroll
  for (int t = 0; t < 4; ++t) {
    int col = colBase + t * 16 + nl;
    float bb2 = b2[col], bbr = bres[col];
#pragma unroll
    for (int r = 0; r < 8; ++r) {
      int row = rowBase + half * 8 + r;
      float di = dinv[row];
      float c1 = acc1[t][r], c2 = acc2[t][r];
      msg2[(size_t)row * N + col] = c1;
      h2[(size_t)row * N + col]   = c2 + bbr + c1 * di * di + bb2;
    }
  }
}

// ---------- GEMM3: a1 = relu(h2@Wa1 + ba1) ----------
__global__ __launch_bounds__(128)
void gemm3_kernel(const unsigned short* __restrict__ A, const unsigned short* __restrict__ Bp,
                  const float* __restrict__ bias, float* __restrict__ out,
                  int M, int K, int N) {
  int lane = threadIdx.x & 31;
  int wave = blockIdx.x * (blockDim.x >> 5) + (threadIdx.x >> 5);
  int nGroups = N >> 6, nTilesTot = N >> 4;
  int totalWaves = (M >> 4) * nGroups;
  if (wave >= totalWaves) return;
  int mt = wave / nGroups, ng = wave % nGroups;
  int rowBase = mt << 4, colBase = ng << 6;
  int half = lane >> 4, nl = lane & 15;
  size_t aRowOff = (size_t)(rowBase + nl) * K;

  v8f acc[4] = {};
  for (int kb = 0; kb < K; kb += 32) {
    v16bf a = load_a_frag_bf(A, aRowOff, kb, half);
    int kt = kb >> 5;
#pragma unroll
    for (int t = 0; t < 4; ++t) {
      v16bf b = load_b_packed(Bp, kt, ng * 4 + t, nTilesTot, lane);
      acc[t] = wmma_bf16(a, b, acc[t]);
    }
  }
#pragma unroll
  for (int t = 0; t < 4; ++t) {
    int col = colBase + t * 16 + nl;
    float bb = bias[col];
#pragma unroll
    for (int r = 0; r < 8; ++r) {
      int row = rowBase + half * 8 + r;
      out[(size_t)row * N + col] = fmaxf(acc[t][r] + bb, 0.0f);
    }
  }
}

// ---------- edge scatter: out[dst] += msg[src] * dinv[src]*dinv[dst] ----------
template <int FSHIFT>
__global__ void scatter_kernel(const float* __restrict__ msg, const int* __restrict__ src,
                               const int* __restrict__ dst, const float* __restrict__ dinv,
                               float* __restrict__ out, int E) {
  long long idx = (long long)blockIdx.x * blockDim.x + threadIdx.x;
  int e = (int)(idx >> FSHIFT);
  int f = (int)(idx & ((1 << FSHIFT) - 1));
  if (e >= E) return;
  int s = src[e], d = dst[e];
  float norm = dinv[s] * dinv[d];
  float v = msg[((size_t)s << FSHIFT) + f] * norm;
  atomicAddF(&out[((size_t)d << FSHIFT) + f], v);
}

// ---------- attention scalar + node counts ----------
__global__ void att_kernel(const float* __restrict__ a1, const float* __restrict__ Wa2,
                           const float* __restrict__ ba2, const int* __restrict__ batch,
                           float* __restrict__ att, float* __restrict__ cnt, int n, int F) {
  int v = blockIdx.x * blockDim.x + threadIdx.x;
  if (v >= n) return;
  const float* row = a1 + (size_t)v * F;
  float acc = ba2[0];
  for (int f = 0; f < F; ++f) acc += row[f] * Wa2[f];
  att[v] = 1.0f / (1.0f + expf(-acc));
  atomicAddF(&cnt[batch[v]], 1.0f);
}

// ---------- weighted pooling ----------
template <int DSHIFT>
__global__ void pool_kernel(const float* __restrict__ h2, const float* __restrict__ att,
                            const int* __restrict__ batch, float* __restrict__ sums, int n) {
  long long idx = (long long)blockIdx.x * blockDim.x + threadIdx.x;
  int v = (int)(idx >> DSHIFT);
  int f = (int)(idx & ((1 << DSHIFT) - 1));
  if (v >= n) return;
  float w = att[v];
  atomicAddF(&sums[((size_t)batch[v] << DSHIFT) + f],
             h2[((size_t)v << DSHIFT) + f] * w);
}

// ---------- finalize: emb = sums/max(cnt,1) ; logits = emb@Wg + bg ----------
__global__ void finalize_kernel(const float* __restrict__ sums, const float* __restrict__ cnt,
                                const float* __restrict__ Wg, const float* __restrict__ bg,
                                float* __restrict__ out, int G, int D, int NG) {
  int tid = threadIdx.x;
  int total = G * D;
  for (int i = tid; i < total; i += blockDim.x) {
    int g = i / D;
    float c = cnt[g];
    c = c > 1.0f ? c : 1.0f;
    out[i] = sums[i] / c;
  }
  __syncthreads();
  int tl = G * NG;
  for (int i = tid; i < tl; i += blockDim.x) {
    int g = i / NG, o = i % NG;
    float acc = bg[o];
    const float* e = out + (size_t)g * D;
    for (int f = 0; f < D; ++f) acc += e[f] * Wg[f * NG + o];
    out[total + i] = acc;
  }
}

// ---------- launch ----------
extern "C" void kernel_launch(void* const* d_in, const int* in_sizes, int n_in,
                              void* d_out, int out_size, void* d_ws, size_t ws_size,
                              hipStream_t stream) {
  (void)n_in; (void)ws_size;
  const float* x    = (const float*)d_in[0];
  const int*   src  = (const int*)d_in[1];
  const int*   dst  = (const int*)d_in[2];
  const int*   batch= (const int*)d_in[3];
  const float* W1   = (const float*)d_in[4];
  const float* b1   = (const float*)d_in[5];
  const float* W2   = (const float*)d_in[6];
  const float* b2   = (const float*)d_in[7];
  const float* Wres = (const float*)d_in[8];
  const float* bres = (const float*)d_in[9];
  const float* Wa1  = (const float*)d_in[10];
  const float* ba1  = (const float*)d_in[11];
  const float* Wa2  = (const float*)d_in[12];
  const float* ba2  = (const float*)d_in[13];
  const float* Wg   = (const float*)d_in[14];
  const float* bg   = (const float*)d_in[15];

  const int Nn  = in_sizes[3];                 // 100000 nodes
  const int E   = in_sizes[1];                 // 1.6M edges
  const int FIN = in_sizes[0] / Nn;            // 512
  const int H   = in_sizes[5];                 // 128
  const int D   = in_sizes[7];                 // 256
  const int A1  = in_sizes[11];                // 64
  const int NG  = in_sizes[15];                // 16
  const int G   = out_size / (D + NG);         // 64

  char* base = (char*)d_ws;
  size_t off = 0;
  auto allocB = [&](size_t bytes) { void* p = base + off; off = (off + bytes + 255) & ~(size_t)255; return p; };
  float* dinv = (float*)allocB(sizeof(float) * Nn);
  float* hw1  = (float*)allocB(sizeof(float) * (size_t)Nn * H);
  float* h1   = (float*)allocB(sizeof(float) * (size_t)Nn * H);
  float* msg2 = (float*)allocB(sizeof(float) * (size_t)Nn * D);
  float* h2   = (float*)allocB(sizeof(float) * (size_t)Nn * D);
  float* a1   = (float*)allocB(sizeof(float) * (size_t)Nn * A1);
  float* att  = (float*)allocB(sizeof(float) * Nn);
  float* sums = (float*)allocB(sizeof(float) * ((size_t)G * D + G));
  float* cnt  = sums + (size_t)G * D;
  unsigned short* h1b   = (unsigned short*)allocB(sizeof(short) * (size_t)Nn * H);
  unsigned short* h2b   = (unsigned short*)allocB(sizeof(short) * (size_t)Nn * D);
  unsigned short* W1p   = (unsigned short*)allocB(sizeof(short) * (size_t)FIN * H);
  unsigned short* W2p   = (unsigned short*)allocB(sizeof(short) * (size_t)H * D);
  unsigned short* Wresp = (unsigned short*)allocB(sizeof(short) * (size_t)H * D);
  unsigned short* Wa1p  = (unsigned short*)allocB(sizeof(short) * (size_t)D * A1);

  auto cdiv = [](long long a, long long b) { return (unsigned)((a + b - 1) / b); };

  // pack weights once into WMMA B-fragment order (bf16)
  pack_b_kernel<<<cdiv((long long)FIN * H, 256), 256, 0, stream>>>(W1, W1p, FIN, H);
  pack_b_kernel<<<cdiv((long long)H * D, 256), 256, 0, stream>>>(W2, W2p, H, D);
  pack_b_kernel<<<cdiv((long long)H * D, 256), 256, 0, stream>>>(Wres, Wresp, H, D);
  pack_b_kernel<<<cdiv((long long)D * A1, 256), 256, 0, stream>>>(Wa1, Wa1p, D, A1);

  // degrees (self-loop = 1) -> dinv
  fill_kernel<<<cdiv(Nn, 256), 256, 0, stream>>>(dinv, 1.0f, Nn);
  deg_edges_kernel<<<cdiv(E, 256), 256, 0, stream>>>(dst, dinv, E);
  rsqrt_kernel<<<cdiv(Nn, 256), 256, 0, stream>>>(dinv, Nn);
  // zero pooling accumulators
  fill_kernel<<<cdiv((long long)G * D + G, 256), 256, 0, stream>>>(sums, 0.0f, (long long)G * D + G);

  // conv1
  gemm1_kernel<<<cdiv((long long)(Nn >> 4) * (H >> 6), 4), 128, 0, stream>>>(
      x, W1p, b1, dinv, hw1, h1, Nn, FIN, H);
  scatter_kernel<7><<<cdiv((long long)E << 7, 256), 256, 0, stream>>>(hw1, src, dst, dinv, h1, E);
  relu_cvt_kernel<<<cdiv((long long)Nn * H, 256), 256, 0, stream>>>(h1, h1b, (long long)Nn * H);

  // conv2 + residual
  gemm2_kernel<<<cdiv((long long)(Nn >> 4) * (D >> 6), 4), 128, 0, stream>>>(
      h1b, W2p, Wresp, b2, bres, dinv, msg2, h2, Nn, H, D);
  scatter_kernel<8><<<cdiv((long long)E << 8, 256), 256, 0, stream>>>(msg2, src, dst, dinv, h2, E);
  relu_inplace_cvt_kernel<<<cdiv((long long)Nn * D, 256), 256, 0, stream>>>(h2, h2b, (long long)Nn * D);

  // attention + pooling
  gemm3_kernel<<<cdiv((long long)(Nn >> 4) * (A1 >> 6), 4), 128, 0, stream>>>(
      h2b, Wa1p, ba1, a1, Nn, D, A1);
  att_kernel<<<cdiv(Nn, 256), 256, 0, stream>>>(a1, Wa2, ba2, batch, att, cnt, Nn, A1);
  pool_kernel<8><<<cdiv((long long)Nn << 8, 256), 256, 0, stream>>>(h2, att, batch, sums, Nn);

  finalize_kernel<<<1, 256, 0, stream>>>(sums, cnt, Wg, bg, (float*)d_out, G, D, NG);
}
